// DAC_85564338471335
// MI455X (gfx1250) — matrix operand order repaired and checked by reading, not compiled, not run on hardware
//
#include <hip/hip_runtime.h>
#include <hip/hip_bf16.h>
#include <cstdint>

typedef _Float16 h16;
typedef __attribute__((ext_vector_type(8)))  _Float16 v8h;
typedef __attribute__((ext_vector_type(16))) _Float16 v16h;
typedef __attribute__((ext_vector_type(8)))  float    v8f;

#define WAVES 4
#define NSUB  2
#define NTILE 128           // WAVES * NSUB * 16 output columns per block
#define BT    128           // 4 wave32 waves per block

__device__ __forceinline__ float snakef(float x, float a) {
  float s = __sinf(a * x);
  return x + s * s / (a + 1e-9f);
}

// ---------------------------------------------------------------------------
// Generic fused conv1d as implicit GEMM on WMMA f16->f32.
//   out[co,n] = bias[co] + (resid? resid[co,n]:0)
//            + sum_{ci,t} W[co,ci,t] * snake(X[ci, n*stride - pad + t*dil])
// X/resid/Out are f16 activations; W/bias/alpha are f32 params.
// K is a template constant so k->(ci,t) folds into mul/shift sequences.
// ---------------------------------------------------------------------------
template <int K>
__global__ void __launch_bounds__(BT) conv_wmma_kernel(
    const h16* __restrict__ X, const float* __restrict__ W,
    const float* __restrict__ bias, const float* __restrict__ alpha,
    const h16* __restrict__ resid, h16* __restrict__ Out,
    int Cin, int Lin, int Lout, int stride, int dil, int pad,
    int chunk, int span)
{
  extern __shared__ h16 smem[];
  h16* Blds = smem;                                   // [chunk][span] activations
  const int boff = (chunk * span + 7) & ~7;           // 16B-align the A region
  h16* Alds = smem + boff;                            // [16][chunk*K] weights

  const int tid  = threadIdx.x;
  const int lane = tid & 31;
  const int wave = tid >> 5;
  const int hh   = lane >> 4;                         // lane half (WMMA layout)
  const int nl   = lane & 15;
  const int n_blk = blockIdx.x * NTILE;
  const int co0   = blockIdx.y * 16;
  const long base_in = (long)n_blk * stride - pad;

  v8f acc[NSUB] = {};

  for (int ci0 = 0; ci0 < Cin; ci0 += chunk) {
    const int cc = (Cin - ci0 < chunk) ? (Cin - ci0) : chunk;
    const int kc = cc * K;                            // guaranteed multiple of 32

    // ---- prefetch head of next chunk's rows (global_prefetch_b8) ----------
    if (ci0 + chunk < Cin) {
      int ncc = (Cin - ci0 - chunk < chunk) ? (Cin - ci0 - chunk) : chunk;
      long g0 = base_in < 0 ? 0 : base_in;
      if (g0 < Lin) {
        for (int ci = tid; ci < ncc; ci += BT)
          __builtin_prefetch(&X[(size_t)(ci0 + chunk + ci) * Lin + g0], 0, 0);
      }
    }

    // ---- stage snake(X) chunk into LDS as f16 (zero pad = post-act zero) --
    const int totB = cc * span;
    for (int i = tid; i < totB; i += BT) {
      int ci  = i / span;
      int col = i - ci * span;
      long g  = base_in + col;
      float v = 0.f;
      if (g >= 0 && g < Lin) {
        v = (float)X[(size_t)(ci0 + ci) * Lin + g];
        if (alpha) v = snakef(v, alpha[ci0 + ci]);
      }
      Blds[i] = (h16)v;
    }
    // ---- stage 16 x (cc*K) weight tile into LDS as f16 --------------------
    const float* Wb = W + (size_t)co0 * Cin * K + (size_t)ci0 * K;
    const int totA = 16 * kc;
    for (int i = tid; i < totA; i += BT) {
      int mrow = i / kc;
      int kk   = i - mrow * kc;
      Alds[i] = (h16)Wb[(size_t)mrow * Cin * K + kk];
    }
    __syncthreads();

    for (int kk = 0; kk < kc; kk += 32) {
      // A fragment (16x32): lane = M row.  Lane's halves are two contiguous
      // 16B-aligned runs -> two ds_load_b128.
      const v8h* arow = (const v8h*)(Alds + nl * kc + kk);   // 16B aligned
      v8h alo = arow[hh];        // k = kk + 8h .. kk + 8h+7
      v8h ahi = arow[2 + hh];    // k = kk + 16+8h .. kk + 23+8h
      v16h afrag = __builtin_shufflevector(alo, ahi,
          0, 1, 2, 3, 4, 5, 6, 7, 8, 9, 10, 11, 12, 13, 14, 15);
#pragma unroll
      for (int s = 0; s < NSUB; ++s) {
        const h16* bbase = Blds + ((wave * NSUB + s) * 16 + nl) * stride;
        // B fragment (32x16): lane = N col, element e -> k = kk + (e&7) + 8*half + 16*(e>>3)
        v16h bfrag;
#pragma unroll
        for (int e = 0; e < 16; ++e) {
          int k  = kk + (e & 7) + 8 * hh + 16 * (e >> 3);
          int ci = k / K;
          int t  = k - ci * K;
          bfrag[e] = bbase[ci * span + t * dil];
        }
        acc[s] = __builtin_amdgcn_wmma_f32_16x16x32_f16(
            false, afrag, false, bfrag, (short)0, acc[s], false, false);
      }
    }
    __syncthreads();
  }

  // Epilogue: C/D layout — lane half selects rows 0-7 / 8-15, lane&15 = N.
#pragma unroll
  for (int s = 0; s < NSUB; ++s) {
    int n = n_blk + (wave * NSUB + s) * 16 + nl;
#pragma unroll
    for (int v = 0; v < 8; ++v) {
      int co = co0 + v + 8 * hh;
      float r = acc[s][v] + bias[co];
      size_t o = (size_t)co * Lout + n;
      if (resid) r += (float)resid[o];
      Out[o] = (h16)r;
    }
  }
}

// ---------------------------------------------------------------------------
// conv_in: 1 -> 64 channels, k=7, pad=3 (no activation on input). Memory-bound.
// ---------------------------------------------------------------------------
__global__ void conv_in_kernel(const float* __restrict__ x, const float* __restrict__ w,
                               const float* __restrict__ b, h16* __restrict__ out, int L) {
  int l = blockIdx.x * blockDim.x + threadIdx.x;
  if (l >= L) return;
  float xs[7];
#pragma unroll
  for (int t = 0; t < 7; ++t) {
    long g = (long)l - 3 + t;
    xs[t] = (g >= 0 && g < L) ? x[g] : 0.f;
  }
  for (int co = 0; co < 64; ++co) {
    float a = b[co];
#pragma unroll
    for (int t = 0; t < 7; ++t) a += w[co * 7 + t] * xs[t];
    out[(size_t)co * L + l] = (h16)a;   // coalesced per co row
  }
}

__global__ void h2f_kernel(const h16* __restrict__ in, float* __restrict__ out, int n) {
  int i = blockIdx.x * blockDim.x + threadIdx.x;
  if (i < n) out[i] = (float)in[i];
}

// ---------------------------------------------------------------------------
// One RVQ codebook step. Column-independent: thread t owns column t.
// argmin(dist) == first argmax of normalized dot product (|e_n|^2,|c_n|^2 const).
// ---------------------------------------------------------------------------
__global__ void __launch_bounds__(256) rvq_step_kernel(
    float* __restrict__ resid, const float* __restrict__ w_in,
    const float* __restrict__ b_in, const float* __restrict__ w_out,
    const float* __restrict__ b_out, const float* __restrict__ cb,
    int* __restrict__ codes, int T)
{
  __shared__ float cbn[1024 * 8];
  __shared__ float wi[8 * 128];
  __shared__ float wo[128 * 8];
  __shared__ float bo[128];
  __shared__ float bi[8];
  int tid = threadIdx.x;
  for (int j = tid; j < 1024; j += 256) {
    float v[8], ss = 0.f;
#pragma unroll
    for (int o = 0; o < 8; ++o) { v[o] = cb[j * 8 + o]; ss += v[o] * v[o]; }
    float inv = 1.f / fmaxf(sqrtf(ss), 1e-12f);
#pragma unroll
    for (int o = 0; o < 8; ++o) cbn[j * 8 + o] = v[o] * inv;
  }
  for (int i = tid; i < 1024; i += 256) wi[i] = w_in[i];
  for (int i = tid; i < 1024; i += 256) wo[i] = w_out[i];
  for (int i = tid; i < 128; i += 256) bo[i] = b_out[i];
  if (tid < 8) bi[tid] = b_in[tid];
  __syncthreads();

  int t = blockIdx.x * 256 + tid;
  if (t >= T) return;

  float ze[8];
#pragma unroll
  for (int o = 0; o < 8; ++o) ze[o] = bi[o];
  for (int c = 0; c < 128; ++c) {
    float rv = resid[(size_t)c * T + t];
#pragma unroll
    for (int o = 0; o < 8; ++o) ze[o] += wi[o * 128 + c] * rv;
  }
  float ss = 0.f;
#pragma unroll
  for (int o = 0; o < 8; ++o) ss += ze[o] * ze[o];
  float inv = 1.f / fmaxf(sqrtf(ss), 1e-12f);
  float en[8];
#pragma unroll
  for (int o = 0; o < 8; ++o) en[o] = ze[o] * inv;

  float best = -1e30f; int bidx = 0;
  for (int j = 0; j < 1024; ++j) {
    float d = 0.f;
#pragma unroll
    for (int o = 0; o < 8; ++o) d += en[o] * cbn[j * 8 + o];  // LDS broadcast
    if (d > best) { best = d; bidx = j; }
  }
  codes[t] = bidx;

  float zq[8];
#pragma unroll
  for (int o = 0; o < 8; ++o) zq[o] = cb[bidx * 8 + o];  // raw codebook gather
  for (int c = 0; c < 128; ++c) {
    float outv = bo[c];
#pragma unroll
    for (int o = 0; o < 8; ++o) outv += wo[c * 8 + o] * zq[o];
    resid[(size_t)c * T + t] -= outv;
  }
}

// ---------------------------------------------------------------------------
// Host-side helpers
// ---------------------------------------------------------------------------
static int gcd32(int k) { int a = k, b = 32; while (b) { int t = a % b; a = b; b = t; } return a; }

static void conv_launch(const void* X, const void* W, const void* B, const void* A,
                        const void* R, void* O, int Cin, int Cout, int Lin, int Lout,
                        int K, int stride, int dil, int pad, hipStream_t st)
{
  int span = (NTILE - 1) * stride + (K - 1) * dil + 1;
  int m = 32 / gcd32(K);                 // chunk multiple so chunk*K % 32 == 0
  int unit = 2 * span + 32 * K;          // f16 bytes per ci in LDS (act + wgt)
  int chunk = 61440 / unit;              // keep LDS <= ~60KB
  if (chunk > Cin) chunk = Cin;
  chunk -= chunk % m;
  if (chunk < m) chunk = m;
  int boff = (chunk * span + 7) & ~7;    // alignment pad (halves)
  size_t sh = (size_t)boff * 2 + (size_t)16 * chunk * K * 2;
  dim3 grid(Lout / NTILE, Cout / 16);
  dim3 blk(BT);
  const h16* x = (const h16*)X;
  const float* w = (const float*)W;
  const float* b = (const float*)B;
  const float* a = (const float*)A;
  const h16* r = (const h16*)R;
  h16* o = (h16*)O;
  switch (K) {
    case 1:  conv_wmma_kernel<1><<<grid, blk, sh, st>>>(x, w, b, a, r, o, Cin, Lin, Lout, stride, dil, pad, chunk, span); break;
    case 3:  conv_wmma_kernel<3><<<grid, blk, sh, st>>>(x, w, b, a, r, o, Cin, Lin, Lout, stride, dil, pad, chunk, span); break;
    case 4:  conv_wmma_kernel<4><<<grid, blk, sh, st>>>(x, w, b, a, r, o, Cin, Lin, Lout, stride, dil, pad, chunk, span); break;
    case 7:  conv_wmma_kernel<7><<<grid, blk, sh, st>>>(x, w, b, a, r, o, Cin, Lin, Lout, stride, dil, pad, chunk, span); break;
    case 8:  conv_wmma_kernel<8><<<grid, blk, sh, st>>>(x, w, b, a, r, o, Cin, Lin, Lout, stride, dil, pad, chunk, span); break;
    case 16: conv_wmma_kernel<16><<<grid, blk, sh, st>>>(x, w, b, a, r, o, Cin, Lin, Lout, stride, dil, pad, chunk, span); break;
    default: break;
  }
}

// Input leaf order (jax pytree, depth-first):
// 0 audio | 1 w_in 2 b_in | per block (21): ru{a1,w1,b1,a2,w2,b2}x3, a, w, b
// 87 a_out 88 w_out 89 b_out | 90 vq_w_in 91 vq_b_in 92 vq_w_out 93 vq_b_out 94 cb
extern "C" void kernel_launch(void* const* d_in, const int* in_sizes, int n_in,
                              void* d_out, int out_size, void* d_ws, size_t ws_size,
                              hipStream_t stream)
{
  (void)in_sizes; (void)n_in; (void)out_size; (void)ws_size;
  const float* audio = (const float*)d_in[0];

  h16* bufA = (h16*)d_ws;                           // 64*1M f16 = 134 MB
  h16* bufB = bufA + (size_t)64 * 1048576;          // second ping-pong buffer
  float* zf = (float*)((char*)d_ws + 268435456ull); // f32 RVQ residual (1 MB)

  int L = 1048576;
  conv_in_kernel<<<dim3((L + 255) / 256), dim3(256), 0, stream>>>(
      audio, (const float*)d_in[1], (const float*)d_in[2], bufA, L);

  h16* cur = bufA;
  h16* oth = bufB;
  const int strides[4] = {2, 4, 8, 8};
  const int dils[3] = {1, 3, 9};
  int pi = 3;
  for (int blk = 0; blk < 4; ++blk) {
    int d2 = 64 << blk;
    int s = strides[blk];
    for (int j = 0; j < 3; ++j) {
      int b0 = pi + 6 * j;
      // y = conv7_dil(snake(x))
      conv_launch(cur, d_in[b0 + 1], d_in[b0 + 2], d_in[b0 + 0], nullptr, oth,
                  d2, d2, L, L, 7, 1, dils[j], 3 * dils[j], stream);
      // x = x + conv1(snake(y))   (in-place: pointwise residual, safe)
      conv_launch(oth, d_in[b0 + 4], d_in[b0 + 5], d_in[b0 + 3], cur, cur,
                  d2, d2, L, L, 1, 1, 1, 0, stream);
    }
    int pad = s / 2;                       // ceil(s/2), s even
    int Kd = 2 * s;
    int Lo = (L + 2 * pad - Kd) / s + 1;
    conv_launch(cur, d_in[pi + 19], d_in[pi + 20], d_in[pi + 18], nullptr, oth,
                d2, 2 * d2, L, Lo, Kd, s, 1, pad, stream);
    { h16* t = cur; cur = oth; oth = t; }
    L = Lo;
    pi += 21;
  }

  // conv_out: 1024 -> 128, k3 pad1, snake input.  L == 2048 here.
  conv_launch(cur, d_in[88], d_in[89], d_in[87], nullptr, oth,
              1024, 128, L, L, 3, 1, 1, 1, stream);

  int nz = 128 * L;
  h2f_kernel<<<dim3((nz + 255) / 256), dim3(256), 0, stream>>>(oth, zf, nz);

  const float* vwi = (const float*)d_in[90];
  const float* vbi = (const float*)d_in[91];
  const float* vwo = (const float*)d_in[92];
  const float* vbo = (const float*)d_in[93];
  const float* vcb = (const float*)d_in[94];
  for (int s = 0; s < 18; ++s) {
    rvq_step_kernel<<<dim3((L + 255) / 256), dim3(256), 0, stream>>>(
        zf, vwi + (size_t)s * 8 * 128, vbi + (size_t)s * 8,
        vwo + (size_t)s * 128 * 8, vbo + (size_t)s * 128,
        vcb + (size_t)s * 1024 * 8, (int*)d_out + (size_t)s * L, L);
  }
}
